// GRUModel_73117523247410
// MI455X (gfx1250) — matrix operand order repaired and checked by reading, not compiled, not run on hardware
//
#include <hip/hip_runtime.h>

// MI455X / gfx1250, wave32. Fully-fused 2-layer GRU + FC, one kernel,
// 128 blocks x 1 wave; each wave owns a 16-row batch tile end-to-end.
// Layer 1 uses a K-concatenated WMMA: A = [h1_t | h2_{t-1}], B = [Wih1^T ; Whh1^T].

typedef __attribute__((ext_vector_type(16))) _Float16 v16h;
typedef __attribute__((ext_vector_type(8)))  _Float16 v8h;
typedef __attribute__((ext_vector_type(8)))  float    v8f;
typedef __attribute__((ext_vector_type(4)))  float    f32x4;

#define GRU_B 2048
#define GRU_T 512
#define GRU_F 32
#define GRU_H 16

// Zero-initialized device global: padded B-fragment halves load their zeros
// from here (real loads -> nothing for the compiler to rematerialize as selects).
__device__ float g_zero16[16] = {0.0f};

__device__ __forceinline__ v8f wmma16(v16h a, v16h b, v8f c) {
  // D = A(16x32 f16) x B(32x16 f16) + C(16x16 f32)
  return __builtin_amdgcn_wmma_f32_16x16x32_f16(false, a, false, b, (short)0, c,
                                                false, false);
}

#define L2E 1.4426950408889634f

// gfx1250 has V_TANH_F32 (TRANS op). Use it if the toolchain exposes a builtin;
// otherwise fall back to the exp2-based formulation. Both paths compile-clean.
#if __has_builtin(__builtin_amdgcn_tanhf)
  #define HAVE_NATIVE_TANH 1
  #define FTANH(x) __builtin_amdgcn_tanhf(x)
#elif __has_builtin(__builtin_amdgcn_tanh_f32)
  #define HAVE_NATIVE_TANH 1
  #define FTANH(x) __builtin_amdgcn_tanh_f32(x)
#else
  #define HAVE_NATIVE_TANH 0
#endif

__global__ __launch_bounds__(32)
void gru2_fused_kernel(const float* __restrict__ x,
                       const float* __restrict__ Wih0, const float* __restrict__ Whh0,
                       const float* __restrict__ bih0, const float* __restrict__ bhh0,
                       const float* __restrict__ Wih1, const float* __restrict__ Whh1,
                       const float* __restrict__ bih1, const float* __restrict__ bhh1,
                       const float* __restrict__ fcw,  const float* __restrict__ fcb,
                       float* __restrict__ out)
{
  __shared__ __align__(16) _Float16 lds0[16 * 16];  // layer-0 h repack buffer
  __shared__ __align__(16) _Float16 lds1[16 * 16];  // layer-1 h repack buffer

  const int  lane = threadIdx.x;       // 0..31
  const int  n16  = lane & 15;
  const bool hi   = lane >= 16;
  const int  off8 = hi ? 8 : 0;
  const int  bb   = blockIdx.x * 16;   // batch tile base

  // ---- build one B-fragment lane-row: 8 f32 from p0 + 8 f32 from p1 -> 16 f16 ----
  auto cvtRow16 = [&](const float* p0, const float* p1) -> v16h {
    v16h v;
#pragma unroll
    for (int i = 0; i < 8; ++i) { v[i] = (_Float16)p0[i]; v[8 + i] = (_Float16)p1[i]; }
    return v;
  };

  // B layout (32x16 f16): column N=lane%16; elems 0-7 = K=off8..off8+7,
  //                       elems 8-15 = K=16+off8..16+off8+7.
  v16h Bx0[3], Bh0[3];
#pragma unroll
  for (int g = 0; g < 3; ++g) {
    // layer0 input weights: dense K=F=32
    Bx0[g] = cvtRow16(Wih0 + (size_t)(g * 16 + n16) * 32 + off8, Wih0 + (size_t)(g * 16 + n16) * 32 + 16 + off8);
    // layer0 recurrent: K=16 real, K>=16 zero (elems 8-15 read the zero row)
    Bh0[g] = cvtRow16(Whh0 + (size_t)(g * 16 + n16) * 16 + off8, g_zero16);
  }
  // layer1 concatenated B for r,z: rows K=0..15 = Wih1^T, K=16..31 = Whh1^T
  v16h Bc1r = cvtRow16(Wih1 + (size_t)(0 * 16 + n16) * 16 + off8, Whh1 + (size_t)(0 * 16 + n16) * 16 + off8);
  v16h Bc1z = cvtRow16(Wih1 + (size_t)(1 * 16 + n16) * 16 + off8, Whh1 + (size_t)(1 * 16 + n16) * 16 + off8);
  // layer1 n-gate: complementary half-zeroed fragments sharing the same A
  v16h Bn1x = cvtRow16(Wih1 + (size_t)(2 * 16 + n16) * 16 + off8, g_zero16);
  v16h Bn1h = cvtRow16(g_zero16, Whh1 + (size_t)(2 * 16 + n16) * 16 + off8);

  // ---- per-lane gate biases (column n16) -------------------------------
  // native-tanh path: sigmoid(v+c) = 0.5 + 0.5*tanh(0.5*v + 0.5*c)
  const float crh0 = 0.5f * (bih0[n16] + bhh0[n16]);
  const float czh0 = 0.5f * (bih0[16 + n16] + bhh0[16 + n16]);
  const float crh1 = 0.5f * (bih1[n16] + bhh1[n16]);
  const float czh1 = 0.5f * (bih1[16 + n16] + bhh1[16 + n16]);
  // exp2-fallback path: pre-scaled by -log2(e)
  const float crp0 = -2.0f * L2E * crh0;
  const float czp0 = -2.0f * L2E * czh0;
  const float crp1 = -2.0f * L2E * crh1;
  const float czp1 = -2.0f * L2E * czh1;
  const float bin0 = bih0[32 + n16];
  const float bhn0 = bhh0[32 + n16];
  const float bin1 = bih1[32 + n16];
  const float bhn1 = bhh1[32 + n16];

  // ---- state ----------------------------------------------------------
  v8f  h0acc = {};   // layer0 hidden, f32, C/D layout
  v8f  h1acc = {};   // layer1 hidden, f32, C/D layout
  v16h hA0;          // A-frag {h1_{t-1} | 0}; high 8 halves written ONCE
  v16h Acat;         // A-frag {h1_t | h2_{t-1}} for layer1
#pragma unroll
  for (int i = 0; i < 16; ++i) { hA0[i] = (_Float16)0.0f; Acat[i] = (_Float16)0.0f; }

  // ---- x prefetch (distance 2, register double-buffer) ----------------
  const float* xrow = x + (size_t)(bb + n16) * GRU_T * GRU_F + (hi ? 8 : 0);
  f32x4 xb0[4], xb1[4];
  auto loadx = [&](int t, f32x4 (&buf)[4]) {
    const float* p = xrow + (size_t)t * GRU_F;
    buf[0] = *(const f32x4*)(p);        // K = k0..k0+3
    buf[1] = *(const f32x4*)(p + 4);    // K = k0+4..k0+7
    buf[2] = *(const f32x4*)(p + 16);   // K = k0+16..k0+19
    buf[3] = *(const f32x4*)(p + 20);   // K = k0+20..k0+23
  };
  auto cvtA = [&](const f32x4 (&b)[4]) -> v16h {
    v16h v;
#pragma unroll
    for (int i = 0; i < 4; ++i) {
      v[i]      = (_Float16)b[0][i];
      v[4 + i]  = (_Float16)b[1][i];
      v[8 + i]  = (_Float16)b[2][i];
      v[12 + i] = (_Float16)b[3][i];
    }
    return v;
  };

  // ---- GRU pointwise in D layout (8 elems/lane) ------------------------
  auto gru_point = [&](v8f& hacc, const v8f& dr, const v8f& dz,
                       const v8f& dxn, const v8f& dhn,
                       float crh, float czh, float crp, float czp,
                       float bin, float bhn) {
#pragma unroll
    for (int e = 0; e < 8; ++e) {
#if HAVE_NATIVE_TANH
      float r = __builtin_fmaf(
          0.5f, FTANH(__builtin_fmaf(dr[e], 0.5f, crh)), 0.5f);
      float z = __builtin_fmaf(
          0.5f, FTANH(__builtin_fmaf(dz[e], 0.5f, czh)), 0.5f);
      float s = __builtin_fmaf(r, dhn[e] + bhn, dxn[e] + bin);
      float n = FTANH(s);
#else
      float r = __builtin_amdgcn_rcpf(
          1.0f + __builtin_amdgcn_exp2f(__builtin_fmaf(dr[e], -L2E, crp)));
      float z = __builtin_amdgcn_rcpf(
          1.0f + __builtin_amdgcn_exp2f(__builtin_fmaf(dz[e], -L2E, czp)));
      float s = __builtin_fmaf(r, dhn[e] + bhn, dxn[e] + bin);
      float q = __builtin_amdgcn_rcpf(
          1.0f + __builtin_amdgcn_exp2f(2.0f * L2E * s));
      float n = __builtin_fmaf(-2.0f, q, 1.0f);          // tanh(s)
      (void)crh; (void)czh;
#endif
      hacc[e] = __builtin_fmaf(z, hacc[e] - n, n);
    }
  };
  // D-layout f32 -> 8 live A-halves via LDS (single wave: in-order, no barrier)
  auto repack = [&](_Float16* lds, const v8f& h) -> v8h {
#pragma unroll
    for (int e = 0; e < 8; ++e)
      lds[(e + off8) * 16 + n16] = (_Float16)h[e];
    return *(const v8h*)(lds + n16 * 16 + off8);
  };

  // layer0 step t: consume buf, reissue load for t+2, 6 WMMAs, update h0
  auto stepL0 = [&](int t, f32x4 (&buf)[4]) {
    v16h xf = cvtA(buf);
    if (t + 2 < GRU_T) loadx(t + 2, buf);   // prefetch distance 2
    v8f zc = {};
    v8f dxr = wmma16(xf, Bx0[0], zc);       // x-side: off the h critical path
    v8f dxz = wmma16(xf, Bx0[1], zc);
    v8f dxn = wmma16(xf, Bx0[2], zc);
    v8f dr  = wmma16(hA0, Bh0[0], dxr);
    v8f dz  = wmma16(hA0, Bh0[1], dxz);
    v8f dhn = wmma16(hA0, Bh0[2], zc);
    gru_point(h0acc, dr, dz, dxn, dhn, crh0, czh0, crp0, czp0, bin0, bhn0);
    v8h row = repack(lds0, h0acc);
#pragma unroll
    for (int e = 0; e < 8; ++e) hA0[e] = row[e];
  };
  // layer1 step: single concatenated A operand, 4 WMMAs
  auto stepL1 = [&]() {
    v8f zc = {};
    v8f dr  = wmma16(Acat, Bc1r, zc);       // xr+hr in one WMMA
    v8f dz  = wmma16(Acat, Bc1z, zc);
    v8f dxn = wmma16(Acat, Bn1x, zc);       // K16..31 half of B is zero
    v8f dhn = wmma16(Acat, Bn1h, zc);       // K0..15  half of B is zero
    gru_point(h1acc, dr, dz, dxn, dhn, crh1, czh1, crp1, czp1, bin1, bhn1);
    v8h row = repack(lds1, h1acc);
#pragma unroll
    for (int e = 0; e < 8; ++e) Acat[8 + e] = row[e];
  };
  auto captureL1in = [&]() {                // Acat.low <- h1_{t-1} (pre-update)
#pragma unroll
    for (int e = 0; e < 8; ++e) Acat[e] = hA0[e];
  };

  // ---- skewed main loop ------------------------------------------------
  loadx(0, xb0);
  loadx(1, xb1);
  stepL0(0, xb0);                             // L0 t=0 (reloads xb0 <- t=2)
#pragma clang loop unroll(disable)
  for (int t = 1; t < GRU_T - 1; t += 2) {    // t = 1,3,...,509
    captureL1in(); stepL0(t,     xb1); stepL1();   // L0 t   | L1 t-1
    captureL1in(); stepL0(t + 1, xb0); stepL1();   // L0 t+1 | L1 t
  }
  captureL1in(); stepL0(GRU_T - 1, xb1); stepL1(); // L0 511 | L1 510
  captureL1in(); stepL1();                         // L1 511

  // ---- final FC: out[b] = h2_last[b,:] . fcw + fcb ---------------------
  // h1acc: lane holds h2[m, n16] for m = e + off8 -> reduce over n (16 lanes)
  float w = fcw[n16];
  float acc[8];
#pragma unroll
  for (int e = 0; e < 8; ++e) acc[e] = h1acc[e] * w;
#pragma unroll
  for (int m = 1; m <= 8; m <<= 1) {
#pragma unroll
    for (int e = 0; e < 8; ++e) acc[e] += __shfl_xor(acc[e], m, 32);
  }
  if (n16 == 0) {
    float b0 = fcb[0];
    int base = bb + off8;
#pragma unroll
    for (int e = 0; e < 8; ++e) out[base + e] = acc[e] + b0;
  }
}

extern "C" void kernel_launch(void* const* d_in, const int* in_sizes, int n_in,
                              void* d_out, int out_size, void* d_ws, size_t ws_size,
                              hipStream_t stream) {
  (void)in_sizes; (void)n_in; (void)out_size; (void)d_ws; (void)ws_size;
  const float* x    = (const float*)d_in[0];
  const float* Wih0 = (const float*)d_in[1];
  const float* Whh0 = (const float*)d_in[2];
  const float* bih0 = (const float*)d_in[3];
  const float* bhh0 = (const float*)d_in[4];
  const float* Wih1 = (const float*)d_in[5];
  const float* Whh1 = (const float*)d_in[6];
  const float* bih1 = (const float*)d_in[7];
  const float* bhh1 = (const float*)d_in[8];
  const float* fcw  = (const float*)d_in[9];
  const float* fcb  = (const float*)d_in[10];
  float* out = (float*)d_out;

  gru2_fused_kernel<<<dim3(GRU_B / 16), dim3(32), 0, stream>>>(
      x, Wih0, Whh0, bih0, bhh0, Wih1, Whh1, bih1, bhh1, fcw, fcb, out);
}
